// DynamicAttention_45526653337814
// MI455X (gfx1250) — matrix operand order repaired and checked by reading, not compiled, not run on hardware
//
#include <hip/hip_runtime.h>
#include <cstdint>

typedef float    v8f  __attribute__((ext_vector_type(8)));
typedef float    v2f  __attribute__((ext_vector_type(2)));
typedef _Float16 v16h __attribute__((ext_vector_type(16)));

#define CC     64
#define C8     8
#define BATCH  2
#define NN     9216        // 96*96
#define KCH    64          // keys per LDS chunk
#define NCHUNK (NN / KCH)  // 144
#define VPITCH 144         // bytes per v row in LDS (128B data + 16B pad -> conflict-free b128)
#define KPITCH 10          // floats per k row in LDS (8 + 2 pad -> conflict-free b64)

#if __has_builtin(__builtin_amdgcn_exp2f)
#define EXP2F(x) __builtin_amdgcn_exp2f(x)
#else
#define EXP2F(x) exp2f(x)
#endif

// ---------------------------------------------------------------------------
// CDNA5 async global->LDS copies (ASYNCcnt-tracked), per CDNA5 ISA.
// ---------------------------------------------------------------------------
__device__ __forceinline__ void async_g2l_b128(uint32_t lds_off, const void* g) {
    asm volatile("global_load_async_to_lds_b128 %0, %1, off"
                 :: "v"(lds_off), "v"((unsigned long long)(uintptr_t)g) : "memory");
}
__device__ __forceinline__ void async_g2l_b32(uint32_t lds_off, const void* g) {
    asm volatile("global_load_async_to_lds_b32 %0, %1, off"
                 :: "v"(lds_off), "v"((unsigned long long)(uintptr_t)g) : "memory");
}
__device__ __forceinline__ void wait_async0() {
    asm volatile("s_wait_asynccnt 0x0" ::: "memory");
}

// ---------------------------------------------------------------------------
// Kernel 1: QKV projections. q is pre-scaled by log2(e) so the attention
// kernel runs softmax natively in exp2 domain.
// q: [B,N,8] f32, k: [B,8,N] f32, vh: [B,64,N] f16.
// ---------------------------------------------------------------------------
__global__ __launch_bounds__(256) void qkv_prep(
    const float* __restrict__ x,
    const float* __restrict__ Wq, const float* __restrict__ bq,
    const float* __restrict__ Wk, const float* __restrict__ bk,
    const float* __restrict__ Wv, const float* __restrict__ bv,
    float* __restrict__ qout, float* __restrict__ kout,
    _Float16* __restrict__ vh)
{
    const int nb = NN / 256;
    const int b  = blockIdx.x / nb;
    const int n  = (blockIdx.x % nb) * 256 + threadIdx.x;

    float xv[CC];
#pragma unroll
    for (int c = 0; c < CC; ++c)
        xv[c] = x[((size_t)b * CC + c) * NN + n];

    for (int o = 0; o < C8; ++o) {
        float a = bq[o];
#pragma unroll
        for (int c = 0; c < CC; ++c) a = fmaf(Wq[o * CC + c], xv[c], a);
        qout[((size_t)b * NN + n) * C8 + o] = a * 1.44269504088896340736f; // log2(e)
    }
    for (int o = 0; o < C8; ++o) {
        float a = bk[o];
#pragma unroll
        for (int c = 0; c < CC; ++c) a = fmaf(Wk[o * CC + c], xv[c], a);
        kout[((size_t)b * C8 + o) * NN + n] = a;
    }
    for (int o = 0; o < CC; ++o) {
        float a = bv[o];
#pragma unroll
        for (int c = 0; c < CC; ++c) a = fmaf(Wv[o * CC + c], xv[c], a);
        vh[((size_t)b * CC + o) * NN + n] = (_Float16)a;
    }
}

__device__ __forceinline__ uint32_t pack2h(float a, float b) {
    union { _Float16 h[2]; uint32_t u; } t;
    t.h[0] = (_Float16)a; t.h[1] = (_Float16)b;
    return t.u;
}

// ---------------------------------------------------------------------------
// Kernel 2: fused flash attention; k/v double-buffered in LDS via async DMA.
// Online-softmax state (m, ssum, acc rescale) updated ONCE per 64-key chunk.
// ---------------------------------------------------------------------------
__global__ __launch_bounds__(256) void attn_flash(
    const float* __restrict__ q,     // [B,N,8] (log2e-scaled)
    const float* __restrict__ k,     // [B,8,N]
    const _Float16* __restrict__ vh, // [B,64,N]
    const float* __restrict__ x,     // [B,64,N]
    const float* __restrict__ gamma, const float* __restrict__ dyn,
    float* __restrict__ out)         // [B,64,N]
{
    __shared__ __align__(16) unsigned char sv[2][CC * VPITCH];   // v chunk: [c][64 keys] f16
    __shared__ __align__(16) float         sk[2][KCH * KPITCH];  // k chunk: [j][8 o] f32 (transposed)

    const int t    = threadIdx.x;
    const int lane = t & 31;
    const int wave = t >> 5;
    const int half = lane >> 4;
    const int ln   = lane & 15;

    const int rowTiles = NN / 128;
    const int b = blockIdx.x / rowTiles;
    const int i = (blockIdx.x % rowTiles) * 128 + wave * 16 + ln; // query row

    const float*         kb  = k + (size_t)b * C8 * NN;
    const unsigned char* vbb = (const unsigned char*)(vh + (size_t)b * CC * NN);

    // q as B-operand of the energy WMMA (32-bit 4x16 layout):
    // lanes<16 carry K={0,1}/{4,5}, lanes>=16 carry K={2,3}/{6,7}.
    const int ob = half * 2;
    const float* qrow = q + ((size_t)b * NN + i) * C8;
    v2f qB0, qB1;
    qB0.x = qrow[ob];     qB0.y = qrow[ob + 1];
    qB1.x = qrow[ob + 4]; qB1.y = qrow[ob + 5];

    v8f acc[4];
#pragma unroll
    for (int ct = 0; ct < 4; ++ct)
        acc[ct] = (v8f){0.f,0.f,0.f,0.f,0.f,0.f,0.f,0.f};
    float m = -3.0e38f, ssum = 0.0f;

    // cooperative async fill of one 64-key chunk (v: 512 b128, k: 512 b32)
    auto load_chunk = [&](int j0, unsigned char* svb, float* skb) {
#pragma unroll
        for (int r = 0; r < 2; ++r) {
            const int m2 = t + r * 256;            // 0..511
            const int c = m2 >> 3, col = m2 & 7;   // 8x16B per v row
            async_g2l_b128((uint32_t)(uintptr_t)(svb + c * VPITCH + col * 16),
                           vbb + (size_t)c * (NN * 2) + (size_t)j0 * 2 + col * 16);
        }
#pragma unroll
        for (int r = 0; r < 2; ++r) {
            const int m2 = t + r * 256;            // 0..511
            const int o = m2 & 7, j = m2 >> 3;     // transpose k on the fly
            async_g2l_b32((uint32_t)(uintptr_t)(skb + j * KPITCH + o),
                          kb + (size_t)o * NN + j0 + j);
        }
    };

    load_chunk(0, sv[0], sk[0]);

    for (int ch = 0; ch < NCHUNK; ++ch) {
        const int cur = ch & 1;
        wait_async0();          // own async writes for chunk `ch` complete
        __syncthreads();        // everyone's writes visible
        if (ch + 1 < NCHUNK)
            load_chunk((ch + 1) * KCH, sv[cur ^ 1], sk[cur ^ 1]);

        const float*         skc = sk[cur];
        const unsigned char* svc = sv[cur];

        // ---- E^T for all 4 key tiles of the chunk (K=8 via chained K=4) ----
        v8f E[4];
#pragma unroll
        for (int tj = 0; tj < 4; ++tj) {
            const float* kr = skc + (tj * 16 + ln) * KPITCH + ob;
            v2f a0 = *(const v2f*)kr;
            v2f a1 = *(const v2f*)(kr + 4);
            v8f e = (v8f){0.f,0.f,0.f,0.f,0.f,0.f,0.f,0.f};
            e = __builtin_amdgcn_wmma_f32_16x16x4_f32(false, a0, false, qB0, (short)0, e, false, false);
            e = __builtin_amdgcn_wmma_f32_16x16x4_f32(false, a1, false, qB1, (short)0, e, false, false);
            E[tj] = e;
        }

        // ---- one online-softmax update per 64 keys ----
        float mx = E[0][0];
#pragma unroll
        for (int tj = 0; tj < 4; ++tj)
#pragma unroll
            for (int g = 0; g < 8; ++g) mx = fmaxf(mx, E[tj][g]);
        mx = fmaxf(mx, __shfl_xor(mx, 16, 32));
        const float mnew  = fmaxf(m, mx);
        const float alpha = EXP2F(m - mnew);
        m = mnew;
#pragma unroll
        for (int ct = 0; ct < 4; ++ct)
#pragma unroll
            for (int g = 0; g < 8; ++g) acc[ct][g] *= alpha;

        float sacc = 0.0f;

        // ---- two 32-key PV halves; E[2s]/E[2s+1] consumed before next ----
#pragma unroll
        for (int s = 0; s < 2; ++s) {
            float p0[8], p1[8];
#pragma unroll
            for (int g = 0; g < 8; ++g) {
                p0[g] = EXP2F(E[2 * s][g] - mnew);
                p1[g] = EXP2F(E[2 * s + 1][g] - mnew);
                sacc += p0[g] + p1[g];
            }

            // P -> f16 B-operand: pack own pairs, one xor-16 swap with
            // half-dependent payload (lower lanes export p1, upper p0).
            uint32_t P0pk[4], P1pk[4], sw[4];
#pragma unroll
            for (int g = 0; g < 4; ++g) {
                P0pk[g] = pack2h(p0[2 * g], p0[2 * g + 1]);
                P1pk[g] = pack2h(p1[2 * g], p1[2 * g + 1]);
            }
#pragma unroll
            for (int g = 0; g < 4; ++g)
                sw[g] = __shfl_xor(half ? P0pk[g] : P1pk[g], 16, 32);

            union { v16h v; uint32_t u[8]; } Bp;
#pragma unroll
            for (int g = 0; g < 4; ++g) {
                Bp.u[g]     = half ? sw[g]   : P0pk[g];  // K-offsets 0..7
                Bp.u[4 + g] = half ? P1pk[g] : sw[g];    // K-offsets 8..15
            }

            // outT(c,i) += V(c,j) * P^T(j,i): 4 c-tiles, f16 K=32
            const unsigned char* vrbase = svc + (s * 32 + half * 16) * 2;
#pragma unroll
            for (int ct = 0; ct < 4; ++ct) {
                const unsigned char* vr = vrbase + (size_t)(ct * 16 + ln) * VPITCH;
                const uint4 A_lo = *(const uint4*)vr;
                const uint4 A_hi = *(const uint4*)(vr + 16);
                union { v16h v; uint32_t u[8]; } Av;
                Av.u[0] = A_lo.x; Av.u[1] = A_lo.y; Av.u[2] = A_lo.z; Av.u[3] = A_lo.w;
                Av.u[4] = A_hi.x; Av.u[5] = A_hi.y; Av.u[6] = A_hi.z; Av.u[7] = A_hi.w;
                acc[ct] = __builtin_amdgcn_wmma_f32_16x16x32_f16(false, Av.v, false, Bp.v,
                                                                 (short)0, acc[ct], false, false);
            }
        }

        sacc += __shfl_xor(sacc, 16, 32);   // single row-sum exchange per chunk
        ssum = ssum * alpha + sacc;

        __syncthreads();        // all waves done reading buf[cur] before reuse
    }

    // ---- epilogue: out = gamma*dyn*(acc/ssum) + x ----
    const float sc = gamma[0] * dyn[0] / ssum;
    const size_t base = (size_t)b * CC * NN;
#pragma unroll
    for (int ct = 0; ct < 4; ++ct) {
#pragma unroll
        for (int g = 0; g < 8; ++g) {
            const size_t idx = base + (size_t)(ct * 16 + g + half * 8) * NN + i;
            out[idx] = fmaf(sc, acc[ct][g], x[idx]);
        }
    }
}

// ---------------------------------------------------------------------------
extern "C" void kernel_launch(void* const* d_in, const int* in_sizes, int n_in,
                              void* d_out, int out_size, void* d_ws, size_t ws_size,
                              hipStream_t stream) {
    const float* x     = (const float*)d_in[0];
    const float* Wq    = (const float*)d_in[1];
    const float* bq    = (const float*)d_in[2];
    const float* Wk    = (const float*)d_in[3];
    const float* bk    = (const float*)d_in[4];
    const float* Wv    = (const float*)d_in[5];
    const float* bv    = (const float*)d_in[6];
    const float* gamma = (const float*)d_in[7];
    const float* dyn   = (const float*)d_in[8];
    float* out = (float*)d_out;

    // workspace layout (16B-aligned slices): vh | q | k
    char* ws = (char*)d_ws;
    _Float16* vh = (_Float16*)ws;                                  // 2*64*9216*2 B
    float* qbuf  = (float*)(ws + (size_t)BATCH * CC * NN * 2);     // 2*9216*8*4  B
    float* kbuf  = qbuf + (size_t)BATCH * NN * C8;                 // 2*8*9216*4  B

    qkv_prep<<<BATCH * (NN / 256), 256, 0, stream>>>(
        x, Wq, bq, Wk, bk, Wv, bv, qbuf, kbuf, vh);

    attn_flash<<<BATCH * (NN / 128), 256, 0, stream>>>(
        qbuf, kbuf, vh, x, gamma, dyn, out);
}
